// GINModel_1013612282298
// MI455X (gfx1250) — matrix operand order repaired and checked by reading, not compiled, not run on hardware
//
#include <hip/hip_runtime.h>
#include <hip/hip_bf16.h>

#define EMB 128
#define HID 128

typedef __bf16 bf16_t;
typedef bf16_t bf16x16 __attribute__((ext_vector_type(16)));
typedef bf16_t bf16x8  __attribute__((ext_vector_type(8)));
typedef float  f32x8   __attribute__((ext_vector_type(8)));

union Frag16 {
    bf16x16 v;
    bf16x8  h[2];
};

static __device__ __forceinline__ void pack8(bf16x16& v, int off, float4 a, float4 b) {
    v[off + 0] = (bf16_t)a.x; v[off + 1] = (bf16_t)a.y;
    v[off + 2] = (bf16_t)a.z; v[off + 3] = (bf16_t)a.w;
    v[off + 4] = (bf16_t)b.x; v[off + 5] = (bf16_t)b.y;
    v[off + 6] = (bf16_t)b.z; v[off + 7] = (bf16_t)b.w;
}

// ---------------------------------------------------------------------------
// x[node, :] = emb[x_idx[node], :]   (float4 per thread, 32 threads/node)
// ---------------------------------------------------------------------------
__global__ __launch_bounds__(256) void embed_kernel(const int* __restrict__ idx,
                                                    const float* __restrict__ emb,
                                                    float* __restrict__ x, int n) {
    int tid = blockIdx.x * 256 + threadIdx.x;
    int node = tid >> 5;
    if (node >= n) return;
    int f = (tid & 31) << 2;
    int e = idx[node];
    *(float4*)(x + (size_t)node * EMB + f) =
        *(const float4*)(emb + (size_t)e * EMB + f);
}

__global__ __launch_bounds__(256) void copy_kernel(const float* __restrict__ src,
                                                   float* __restrict__ dst, long n4) {
    long i = (long)blockIdx.x * 256 + threadIdx.x;
    if (i < n4) ((float4*)dst)[i] = ((const float4*)src)[i];
}

__global__ __launch_bounds__(256) void zero_kernel(float* __restrict__ p, int n) {
    int i = blockIdx.x * 256 + threadIdx.x;
    if (i < n) p[i] = 0.0f;
}

// ---------------------------------------------------------------------------
// agg[dst[e], :] += x[src[e], :]   (float4 read, 4x global_atomic_add_f32)
// ---------------------------------------------------------------------------
__global__ __launch_bounds__(256) void edge_scatter_kernel(const int* __restrict__ src,
                                                           const int* __restrict__ dst,
                                                           const float* __restrict__ x,
                                                           float* __restrict__ agg,
                                                           int n_edges) {
    int tid = blockIdx.x * 256 + threadIdx.x;
    int e = tid >> 5;
    if (e >= n_edges) return;
    int f = (tid & 31) << 2;
    int s = src[e], d = dst[e];
    float4 v = *(const float4*)(x + (size_t)s * HID + f);
    float* ap = agg + (size_t)d * HID + f;
    atomicAdd(ap + 0, v.x);
    atomicAdd(ap + 1, v.y);
    atomicAdd(ap + 2, v.z);
    atomicAdd(ap + 3, v.w);
}

// ---------------------------------------------------------------------------
// Fused GIN MLP: xout = relu( relu((x+agg) @ W1 + b1) @ W2 + b2 )
// 256 threads = 8 waves; each wave computes a 16-row node tile with
// v_wmma_f32_16x16x32_bf16 (8 N-tiles x 4 K-chunks = 32 WMMAs per linear).
//
// Weight B-fragment layout in LDS (per ISA 16-bit B, 32x16 tile):
//   lane l: column n = 16*t + (l&15); elem e holds K = c*32 + (l>>4)*8 + (e&7) + (e>>3)*16
//   stored so each lane's v16bf is 32 contiguous bytes -> ds_load_b128 pairs.
// A-fragment (per ISA 16-bit A, 16x32 tile):
//   lane l: row m = l&15; VGPRs0-3: K = hk..hk+7, VGPRs4-7: K = 16+hk..hk+23,
//   hk = (l>>4)*8  -> two contiguous 8-float runs -> float4 global loads.
// ---------------------------------------------------------------------------
__global__ __launch_bounds__(256) void gin_mlp_kernel(const float* __restrict__ xin,
                                                      const float* __restrict__ w1,
                                                      const float* __restrict__ b1,
                                                      const float* __restrict__ w2,
                                                      const float* __restrict__ b2,
                                                      float* __restrict__ xout,
                                                      int n_nodes) {
    __shared__ __attribute__((aligned(32))) bf16_t w1s[128 * 128];
    __shared__ __attribute__((aligned(32))) bf16_t w2s[128 * 128];
    __shared__ __attribute__((aligned(32))) bf16_t h1s[8][16 * 128];
    __shared__ float b1s[128], b2s[128];

    const int tid = threadIdx.x;
    const int lane = tid & 31;
    const int wave = tid >> 5;

    // ---- load + swizzle both weight matrices into fragment layout ----
    // frag id f = t*4 + c ; LDS element ((f*32+ln)*16 + e) <- W[k][n],
//   n = t*16 + (ln&15), k = c*32 + (ln>>4)*8 + (e&7) + (e>>3)*16
    for (int i = tid; i < 1024; i += 256) {
        int fg = i >> 5, ln = i & 31;
        int t = fg >> 2, c = fg & 3;
        int n = t * 16 + (ln & 15);
        int kbase = c * 32 + (ln >> 4) * 8;
        int base = i * 16;
#pragma unroll
        for (int e = 0; e < 16; ++e) {
            int k = kbase + (e & 7) + ((e >> 3) << 4);
            w1s[base + e] = (bf16_t)w1[k * HID + n];
            w2s[base + e] = (bf16_t)w2[k * HID + n];
        }
    }
    if (tid < 128) { b1s[tid] = b1[tid]; b2s[tid] = b2[tid]; }
    __syncthreads();

    const int row0 = blockIdx.x * 128 + wave * 16;
    const int m  = lane & 15;
    const int hk = (lane >> 4) << 3;   // 0 or 8
    const int nn = lane & 15;
    const int rb = (lane >> 4) << 3;   // C/D: rows r+0..7 or r+8..15

    // ---- A fragments from global (x + agg), f32 -> bf16 ----
    int rowm = row0 + m;
    const float* arow = xin + (size_t)(rowm < n_nodes ? rowm : n_nodes - 1) * HID;
    Frag16 a[4];
#pragma unroll
    for (int c = 0; c < 4; ++c) {
        const float* p = arow + c * 32 + hk;
        pack8(a[c].v, 0, *(const float4*)(p),      *(const float4*)(p + 4));
        pack8(a[c].v, 8, *(const float4*)(p + 16), *(const float4*)(p + 20));
    }

    // ---- GEMM 1: C1 = A @ W1 ----
    f32x8 acc[8] = {};
#pragma unroll
    for (int c = 0; c < 4; ++c) {
#pragma unroll
        for (int t = 0; t < 8; ++t) {
            bf16x16 bf = *(const bf16x16*)&w1s[((t * 4 + c) * 32 + lane) * 16];
            acc[t] = __builtin_amdgcn_wmma_f32_16x16x32_bf16(
                false, a[c].v, false, bf, (short)0, acc[t], false, false);
        }
    }

    // ---- bias + relu, store H1 (bf16, row-major) to per-wave LDS tile ----
    bf16_t* hrow = &h1s[wave][0];
#pragma unroll
    for (int t = 0; t < 8; ++t) {
        float bv = b1s[t * 16 + nn];
#pragma unroll
        for (int r = 0; r < 8; ++r) {
            float hv = acc[t][r] + bv;
            hrow[(r + rb) * HID + t * 16 + nn] = (bf16_t)fmaxf(hv, 0.0f);
        }
    }

    // ---- reload H1 as A fragments (two contiguous 16B LDS reads each) ----
    Frag16 a2[4];
    const bf16_t* hr = hrow + m * HID;
#pragma unroll
    for (int c = 0; c < 4; ++c) {
        a2[c].h[0] = *(const bf16x8*)(hr + c * 32 + hk);
        a2[c].h[1] = *(const bf16x8*)(hr + c * 32 + 16 + hk);
    }

    // ---- GEMM 2: C2 = H1 @ W2 ----
    f32x8 acc2[8] = {};
#pragma unroll
    for (int c = 0; c < 4; ++c) {
#pragma unroll
        for (int t = 0; t < 8; ++t) {
            bf16x16 bf = *(const bf16x16*)&w2s[((t * 4 + c) * 32 + lane) * 16];
            acc2[t] = __builtin_amdgcn_wmma_f32_16x16x32_bf16(
                false, a2[c].v, false, bf, (short)0, acc2[t], false, false);
        }
    }

    // ---- bias + relu, store x_out (f32) ----
#pragma unroll
    for (int t = 0; t < 8; ++t) {
        float bv = b2s[t * 16 + nn];
#pragma unroll
        for (int r = 0; r < 8; ++r) {
            int node = row0 + r + rb;
            if (node < n_nodes)
                xout[(size_t)node * HID + t * 16 + nn] = fmaxf(acc2[t][r] + bv, 0.0f);
        }
    }
}

// ---------------------------------------------------------------------------
// Segment-sum pooling via atomics (batch ids; counts from lane 0)
// ---------------------------------------------------------------------------
__global__ __launch_bounds__(256) void pool_kernel(const int* __restrict__ batch,
                                                   const float* __restrict__ x,
                                                   float* __restrict__ sums,
                                                   float* __restrict__ cnts, int n) {
    int tid = blockIdx.x * 256 + threadIdx.x;
    int node = tid >> 5;
    if (node >= n) return;
    int f = (tid & 31) << 2;
    int b = batch[node];
    float4 v = *(const float4*)(x + (size_t)node * HID + f);
    float* sp = sums + (size_t)b * HID + f;
    atomicAdd(sp + 0, v.x);
    atomicAdd(sp + 1, v.y);
    atomicAdd(sp + 2, v.z);
    atomicAdd(sp + 3, v.w);
    if ((tid & 31) == 0) atomicAdd(cnts + b, 1.0f);
}

// ---------------------------------------------------------------------------
// Per-graph classifier: relu(g @ W1 + b1) @ W2 + b2 -> softmax over 2 classes
// ---------------------------------------------------------------------------
__global__ __launch_bounds__(64) void classifier_kernel(const float* __restrict__ sums,
                                                        const float* __restrict__ cnts,
                                                        const float* __restrict__ w1,
                                                        const float* __restrict__ b1,
                                                        const float* __restrict__ w2,
                                                        const float* __restrict__ b2,
                                                        float* __restrict__ out) {
    __shared__ float g[128];
    __shared__ float h[64];
    __shared__ float z[2];
    int b = blockIdx.x, j = threadIdx.x;
    float inv = 1.0f / fmaxf(cnts[b], 1.0f);
    g[j]      = sums[(size_t)b * HID + j] * inv;
    g[j + 64] = sums[(size_t)b * HID + 64 + j] * inv;
    __syncthreads();
    float acc = b1[j];
#pragma unroll 8
    for (int f = 0; f < 128; ++f) acc += g[f] * w1[f * 64 + j];
    h[j] = fmaxf(acc, 0.0f);
    __syncthreads();
    if (j < 2) {
        float zz = b2[j];
        for (int k = 0; k < 64; ++k) zz += h[k] * w2[k * 2 + j];
        z[j] = zz;
    }
    __syncthreads();
    if (j == 0) {
        float mz = fmaxf(z[0], z[1]);
        float e0 = __expf(z[0] - mz), e1 = __expf(z[1] - mz);
        float s = e0 + e1;
        out[(size_t)b * 2 + 0] = e0 / s;
        out[(size_t)b * 2 + 1] = e1 / s;
    }
}

// ---------------------------------------------------------------------------
extern "C" void kernel_launch(void* const* d_in, const int* in_sizes, int n_in,
                              void* d_out, int out_size, void* d_ws, size_t ws_size,
                              hipStream_t stream) {
    const int*   x_idx = (const int*)d_in[0];
    const int*   eidx  = (const int*)d_in[1];
    const int*   batch = (const int*)d_in[2];
    const float* emb   = (const float*)d_in[3];
    const float* cw1   = (const float*)d_in[4];
    const float* cb1   = (const float*)d_in[5];
    const float* cw2   = (const float*)d_in[6];
    const float* cb2   = (const float*)d_in[7];
    const float* mw1   = (const float*)d_in[8];
    const float* mb1   = (const float*)d_in[9];
    const float* mw2   = (const float*)d_in[10];
    const float* mb2   = (const float*)d_in[11];

    const int n_nodes  = in_sizes[0];
    const int n_edges  = in_sizes[1] / 2;
    const int n_graphs = out_size / 2;

    const int* src = eidx;
    const int* dst = eidx + n_edges;

    // workspace: x [N,128] f32 | agg [N,128] f32 (sums/cnts alias agg afterwards)
    float* x    = (float*)d_ws;
    float* agg  = x + (size_t)n_nodes * HID;
    float* sums = agg;                               // agg is dead after layer 3
    float* cnts = sums + (size_t)n_graphs * HID;

    dim3 b256(256);
    const int nodeThreads = n_nodes * 32;
    const int edgeThreads = n_edges * 32;
    const long n4 = (long)n_nodes * HID / 4;
    const int mlpBlocks = (n_nodes + 127) / 128;

    embed_kernel<<<(nodeThreads + 255) / 256, b256, 0, stream>>>(x_idx, emb, x, n_nodes);

    for (int l = 0; l < 3; ++l) {
        copy_kernel<<<(int)((n4 + 255) / 256), b256, 0, stream>>>(x, agg, n4);
        edge_scatter_kernel<<<(edgeThreads + 255) / 256, b256, 0, stream>>>(
            src, dst, x, agg, n_edges);
        gin_mlp_kernel<<<mlpBlocks, b256, 0, stream>>>(
            agg,
            cw1 + (size_t)l * HID * HID, cb1 + (size_t)l * HID,
            cw2 + (size_t)l * HID * HID, cb2 + (size_t)l * HID,
            x, n_nodes);
    }

    const int nz = n_graphs * HID + n_graphs;
    zero_kernel<<<(nz + 255) / 256, b256, 0, stream>>>(sums, nz);
    pool_kernel<<<(nodeThreads + 255) / 256, b256, 0, stream>>>(batch, x, sums, cnts, n_nodes);
    classifier_kernel<<<n_graphs, 64, 0, stream>>>(sums, cnts, mw1, mb1, mw2, mb2,
                                                   (float*)d_out);
}
